// RelationalTransformer_23983097381597
// MI455X (gfx1250) — compile-verified, matrix operand order checked by reading, not run on hardware
//
#include <hip/hip_runtime.h>

typedef __bf16 bf16_t;
typedef __attribute__((ext_vector_type(16))) __bf16 v16bf;
typedef __attribute__((ext_vector_type(8)))  float  v8f;

#define N_  10000
#define E_  160000
#define D_  128
#define H_  4
#define FF_ 512
#define L_  2

__device__ __forceinline__ v8f zero8() {
  v8f z = {0.f,0.f,0.f,0.f,0.f,0.f,0.f,0.f};
  return z;
}

__device__ __forceinline__ v8f wmma_bf16(v16bf a, v16bf b, v8f c) {
  return __builtin_amdgcn_wmma_f32_16x16x32_bf16(false, a, false, b, (short)0, c, false, false);
}

// K index inside a 16-bit A/B fragment element e (0..15), per CDNA5 ISA layout:
// lanes 0-15 hold K {0..7,16..23}; lanes 16-31 hold K {8..15,24..31}
__device__ __forceinline__ int frag_k(int e, int hi) {
  int j = e >> 1;
  return ((j >> 2) << 4) + (hi << 3) + ((j & 3) << 1) + (e & 1);
}

// A fragment: 16x32 tile from row-major bf16 (LDS), rows = M, cols = K
__device__ __forceinline__ v16bf load_a(const bf16_t* s, int ld, int k0, int lane) {
  int m = lane & 15, hi = lane >> 4;
  v16bf a;
#pragma unroll
  for (int e = 0; e < 16; ++e) a[e] = s[m * ld + k0 + frag_k(e, hi)];
  return a;
}

// B fragment: 32x16 tile of W^T where W is row-major [out,in]; B[k][n] = W[n0+n][k0+k]
__device__ __forceinline__ v16bf load_b(const bf16_t* w, int ld, int n0, int k0, int lane) {
  int n = lane & 15, hi = lane >> 4;
  v16bf b;
#pragma unroll
  for (int e = 0; e < 16; ++e) b[e] = w[(n0 + n) * ld + k0 + frag_k(e, hi)];
  return b;
}

__device__ __forceinline__ void atomicMaxF(float* addr, float val) {
  unsigned int* ai = (unsigned int*)addr;
  unsigned int old = *ai;
  while (true) {
    float f = __uint_as_float(old);
    if (f >= val) break;
    unsigned int prev = atomicCAS(ai, old, __float_as_uint(val));
    if (prev == old) break;
    old = prev;
  }
}

// ---------------- utility kernels ----------------
__global__ void k_cvt(const float* __restrict__ in, bf16_t* __restrict__ out, int n) {
  int i = blockIdx.x * 256 + threadIdx.x;
  if (i < n) out[i] = (bf16_t)in[i];
}

__global__ void k_gather_x(const int* __restrict__ ids, const float* __restrict__ emb,
                           float* __restrict__ x, int total) {
  int i = blockIdx.x * 256 + threadIdx.x;
  if (i < total) x[i] = emb[ids[i >> 7] * D_ + (i & 127)];
}

__global__ void k_gather_e(const int* __restrict__ ids, const float* __restrict__ emb,
                           bf16_t* __restrict__ ef, int total) {
  int i = blockIdx.x * 256 + threadIdx.x;
  if (i < total) ef[i] = (bf16_t)emb[ids[i >> 7] * D_ + (i & 127)];
}

__global__ void k_init(float* __restrict__ smax, float* __restrict__ den,
                       float* __restrict__ agg, int NH, int ND) {
  int i = blockIdx.x * 256 + threadIdx.x;
  if (i < ND) agg[i] = 0.f;
  if (i < NH) { smax[i] = -3.0e38f; den[i] = 0.f; }
}

__global__ void k_copy(const float* __restrict__ x, float* __restrict__ o, int n) {
  int i = blockIdx.x * 256 + threadIdx.x;
  if (i < n) o[i] = x[i];
}

// ---------------- q = LN(x) @ q_w^T ----------------
__global__ void k_q(const float* __restrict__ x, const float* __restrict__ lnw,
                    const float* __restrict__ lnb, const bf16_t* __restrict__ qw,
                    float* __restrict__ q) {
  __shared__ bf16_t sa[16 * 136];
  int n0 = blockIdx.x * 16;
  int tid = threadIdx.x, lane = tid & 31, wv = tid >> 5;
  {
    int row = wv * 2 + (lane >> 4);
    int cl = lane & 15;
    const float* xr = x + (n0 + row) * D_;
    float s = 0.f, s2 = 0.f, vals[8];
#pragma unroll
    for (int t = 0; t < 8; ++t) { float v = xr[cl + 16 * t]; vals[t] = v; s += v; s2 += v * v; }
#pragma unroll
    for (int m = 1; m < 16; m <<= 1) { s += __shfl_xor(s, m, 32); s2 += __shfl_xor(s2, m, 32); }
    float mu = s * (1.f / 128.f);
    float var = s2 * (1.f / 128.f) - mu * mu;
    float inv = rsqrtf(var + 1e-6f);
#pragma unroll
    for (int t = 0; t < 8; ++t) {
      int c = cl + 16 * t;
      sa[row * 136 + c] = (bf16_t)((vals[t] - mu) * inv * lnw[c] + lnb[c]);
    }
  }
  __syncthreads();
  v8f acc = zero8();
#pragma unroll
  for (int kk = 0; kk < 4; ++kk) {
    v16bf a = load_a(sa, 136, kk * 32, lane);
    v16bf b = load_b(qw, 128, wv * 16, kk * 32, lane);
    acc = wmma_bf16(a, b, acc);
  }
  int nc = wv * 16 + (lane & 15), hi = lane >> 4;
#pragma unroll
  for (int r = 0; r < 8; ++r) q[(n0 + r + hi * 8) * D_ + nc] = acc[r];
}

// ---------------- fused per-edge pipeline (64 edges / block) ----------------
#define EB     64
#define CAT_LD 272
#define INT_LD 528
#define KV_LD  132
#define SMEM_EDGE (EB * CAT_LD * 2 + EB * INT_LD * 2 + EB * KV_LD * 4)

#define INTER_EPI(ACC, RB) do { \
  int nc_ = tile * 16 + (lane & 15); int hi_ = lane >> 4; \
  float bias_ = ba[nc_]; \
  _Pragma("unroll") \
  for (int r = 0; r < 8; ++r) { \
    float vv = ACC[r] + bias_; \
    vv = vv > 0.f ? vv : (__expf(vv) - 1.f); \
    inter[((RB) * 16 + r + hi_ * 8) * INT_LD + nc_] = (bf16_t)vv; \
  } \
} while (0)

#define GATE_EPI(GACC, BACC, RB) do { \
  int nc_ = wv * 16 + (lane & 15); int hi_ = lane >> 4; \
  float gb_ = bg[nc_], bb_ = bbias[nc_]; \
  _Pragma("unroll") \
  for (int r = 0; r < 8; ++r) { \
    int m_ = (RB) * 16 + r + hi_ * 8; \
    float g_ = 1.f / (1.f + __expf(-(GACC[r] + gb_ + 3.0f))); \
    float kadd_ = BACC[r] + bb_; \
    float kj_ = (float)cat[m_ * CAT_LD + nc_]; \
    cat[m_ * CAT_LD + nc_] = (bf16_t)(kj_ * g_ + kadd_ * (1.f - g_)); \
  } \
} while (0)

#define KV_EPI(KACC, VACC, RB) do { \
  int nc_ = wv * 16 + (lane & 15); int hi_ = lane >> 4; \
  _Pragma("unroll") \
  for (int r = 0; r < 8; ++r) { \
    int m_ = (RB) * 16 + r + hi_ * 8; \
    kS[m_ * KV_LD + nc_] = KACC[r]; \
    vout[(e0 + m_) * D_ + nc_] = (bf16_t)VACC[r]; \
  } \
} while (0)

__global__ void k_edge(const float* __restrict__ x, const bf16_t* __restrict__ ef,
                       const float* __restrict__ q,
                       const int* __restrict__ src, const int* __restrict__ dst,
                       const bf16_t* __restrict__ wa, const float* __restrict__ ba,
                       const bf16_t* __restrict__ wg, const float* __restrict__ bg,
                       const bf16_t* __restrict__ wb, const float* __restrict__ bbias,
                       const bf16_t* __restrict__ wk, const bf16_t* __restrict__ wvv,
                       bf16_t* __restrict__ vout, float* __restrict__ scores,
                       float* __restrict__ smax) {
  extern __shared__ char smem[];
  bf16_t* cat   = (bf16_t*)smem;                                    // EB x CAT_LD
  bf16_t* inter = (bf16_t*)(smem + EB * CAT_LD * 2);                // EB x INT_LD
  float*  kS    = (float*)(smem + EB * CAT_LD * 2 + EB * INT_LD * 2); // EB x KV_LD
  int e0 = blockIdx.x * EB;
  int tid = threadIdx.x, lane = tid & 31, wv = tid >> 5;
  // load cat = [bf16(x[src]), ef] : 4 threads per row, 64 cols each
  {
    int row = tid >> 2;
    int cseg = (tid & 3) * 64;
    int e = e0 + row;
    if (cseg < 128) {
      const float* xr = x + src[e] * D_ + cseg;
#pragma unroll
      for (int t = 0; t < 64; ++t) cat[row * CAT_LD + cseg + t] = (bf16_t)xr[t];
    } else {
      const bf16_t* er = ef + e * D_ + (cseg - 128);
#pragma unroll
      for (int t = 0; t < 64; ++t) cat[row * CAT_LD + cseg + t] = er[t];
    }
  }
  __syncthreads();
  // inter = celu(cat @ wa^T + ba): 32 col tiles x 4 row-blocks; B reused 4x
  for (int tt = 0; tt < 4; ++tt) {
    int tile = wv + tt * 8;
    v8f ac0 = zero8(), ac1 = zero8(), ac2 = zero8(), ac3 = zero8();
#pragma unroll
    for (int kk = 0; kk < 8; ++kk) {
      v16bf b = load_b(wa, 256, tile * 16, kk * 32, lane);
      v16bf a0 = load_a(cat + 0 * 16 * CAT_LD, CAT_LD, kk * 32, lane);
      ac0 = wmma_bf16(a0, b, ac0);
      v16bf a1 = load_a(cat + 1 * 16 * CAT_LD, CAT_LD, kk * 32, lane);
      ac1 = wmma_bf16(a1, b, ac1);
      v16bf a2 = load_a(cat + 2 * 16 * CAT_LD, CAT_LD, kk * 32, lane);
      ac2 = wmma_bf16(a2, b, ac2);
      v16bf a3 = load_a(cat + 3 * 16 * CAT_LD, CAT_LD, kk * 32, lane);
      ac3 = wmma_bf16(a3, b, ac3);
    }
    INTER_EPI(ac0, 0); INTER_EPI(ac1, 1); INTER_EPI(ac2, 2); INTER_EPI(ac3, 3);
  }
  __syncthreads();
  // gate / k_add -> kv2 (in-place into cat); wave owns 16 output cols, 4 row-blocks
  {
    v8f g0 = zero8(), g1 = zero8(), g2 = zero8(), g3 = zero8();
    v8f c0 = zero8(), c1 = zero8(), c2 = zero8(), c3 = zero8();
#pragma unroll
    for (int kk = 0; kk < 16; ++kk) {
      v16bf bG = load_b(wg, 512, wv * 16, kk * 32, lane);
      v16bf bB = load_b(wb, 512, wv * 16, kk * 32, lane);
      v16bf a0 = load_a(inter + 0 * 16 * INT_LD, INT_LD, kk * 32, lane);
      g0 = wmma_bf16(a0, bG, g0); c0 = wmma_bf16(a0, bB, c0);
      v16bf a1 = load_a(inter + 1 * 16 * INT_LD, INT_LD, kk * 32, lane);
      g1 = wmma_bf16(a1, bG, g1); c1 = wmma_bf16(a1, bB, c1);
      v16bf a2 = load_a(inter + 2 * 16 * INT_LD, INT_LD, kk * 32, lane);
      g2 = wmma_bf16(a2, bG, g2); c2 = wmma_bf16(a2, bB, c2);
      v16bf a3 = load_a(inter + 3 * 16 * INT_LD, INT_LD, kk * 32, lane);
      g3 = wmma_bf16(a3, bG, g3); c3 = wmma_bf16(a3, bB, c3);
    }
    GATE_EPI(g0, c0, 0); GATE_EPI(g1, c1, 1); GATE_EPI(g2, c2, 2); GATE_EPI(g3, c3, 3);
  }
  __syncthreads();
  // k = [kv2,ef] @ wk^T (LDS fp32), v = kv2 @ wv^T (global bf16)
  {
    v8f k0 = zero8(), k1 = zero8(), k2 = zero8(), k3 = zero8();
#pragma unroll
    for (int kk = 0; kk < 8; ++kk) {
      v16bf b = load_b(wk, 256, wv * 16, kk * 32, lane);
      v16bf a0 = load_a(cat + 0 * 16 * CAT_LD, CAT_LD, kk * 32, lane);
      k0 = wmma_bf16(a0, b, k0);
      v16bf a1 = load_a(cat + 1 * 16 * CAT_LD, CAT_LD, kk * 32, lane);
      k1 = wmma_bf16(a1, b, k1);
      v16bf a2 = load_a(cat + 2 * 16 * CAT_LD, CAT_LD, kk * 32, lane);
      k2 = wmma_bf16(a2, b, k2);
      v16bf a3 = load_a(cat + 3 * 16 * CAT_LD, CAT_LD, kk * 32, lane);
      k3 = wmma_bf16(a3, b, k3);
    }
    v8f v0 = zero8(), v1 = zero8(), v2 = zero8(), v3 = zero8();
#pragma unroll
    for (int kk = 0; kk < 4; ++kk) {
      v16bf b = load_b(wvv, 128, wv * 16, kk * 32, lane);
      v16bf a0 = load_a(cat + 0 * 16 * CAT_LD, CAT_LD, kk * 32, lane);
      v0 = wmma_bf16(a0, b, v0);
      v16bf a1 = load_a(cat + 1 * 16 * CAT_LD, CAT_LD, kk * 32, lane);
      v1 = wmma_bf16(a1, b, v1);
      v16bf a2 = load_a(cat + 2 * 16 * CAT_LD, CAT_LD, kk * 32, lane);
      v2 = wmma_bf16(a2, b, v2);
      v16bf a3 = load_a(cat + 3 * 16 * CAT_LD, CAT_LD, kk * 32, lane);
      v3 = wmma_bf16(a3, b, v3);
    }
    KV_EPI(k0, v0, 0); KV_EPI(k1, v1, 1); KV_EPI(k2, v2, 2); KV_EPI(k3, v3, 3);
  }
  __syncthreads();
  // scores = sum_c q[dst]*k per head + segment max : one (edge,head) per thread
  {
    int r = tid >> 2, h = tid & 3;
    int e = e0 + r;
    int d = dst[e];
    const float* qr = q + d * D_ + h * 32;
    float s = 0.f;
#pragma unroll
    for (int c = 0; c < 32; ++c) s += qr[c] * kS[r * KV_LD + h * 32 + c];
    scores[e * H_ + h] = s;
    atomicMaxF(&smax[d * H_ + h], s);
  }
}

// ---------------- segment softmax pass 2 ----------------
__global__ void k_soft(const int* __restrict__ dst, float* __restrict__ scores,
                       const float* __restrict__ smax, float* __restrict__ den, int EH) {
  int i = blockIdx.x * 256 + threadIdx.x;
  if (i >= EH) return;
  int e = i >> 2, h = i & 3;
  int d = dst[e];
  float ex = __expf(scores[i] - smax[d * H_ + h]);
  scores[i] = ex;
  atomicAdd(&den[d * H_ + h], ex);
}

// ---------------- weighted scatter of messages ----------------
__global__ void k_scatter(const int* __restrict__ dst, const float* __restrict__ scores,
                          const float* __restrict__ den, const bf16_t* __restrict__ v,
                          float* __restrict__ agg, int total) {
  int i = blockIdx.x * 256 + threadIdx.x;
  if (i >= total) return;
  int e = i >> 7, c = i & 127, h = c >> 5;
  int d = dst[e];
  float alpha = scores[e * H_ + h] / den[d * H_ + h];
  atomicAdd(&agg[d * D_ + c], (float)v[i] * alpha);
}

// ---------------- node update: o-proj residual + FFN ----------------
__global__ void k_node(float* __restrict__ x, const float* __restrict__ agg,
                       const bf16_t* __restrict__ ow,
                       const float* __restrict__ flnw, const float* __restrict__ flnb,
                       const bf16_t* __restrict__ wi, const bf16_t* __restrict__ wo) {
  __shared__ bf16_t sa[16 * 136];
  __shared__ bf16_t h2[16 * 528];
  __shared__ float  xs[16 * 132];
  int n0 = blockIdx.x * 16;
  int tid = threadIdx.x, lane = tid & 31, wv = tid >> 5;
  {
    int row = tid >> 4;
    int cb = (tid & 15) * 8;
    const float* ar = agg + (n0 + row) * D_;
    const float* xr = x + (n0 + row) * D_;
#pragma unroll
    for (int t = 0; t < 8; ++t) {
      sa[row * 136 + cb + t] = (bf16_t)ar[cb + t];
      xs[row * 132 + cb + t] = xr[cb + t];
    }
  }
  __syncthreads();
  // x += agg @ ow^T
  {
    v8f acc = zero8();
#pragma unroll
    for (int kk = 0; kk < 4; ++kk) {
      v16bf a = load_a(sa, 136, kk * 32, lane);
      v16bf b = load_b(ow, 128, wv * 16, kk * 32, lane);
      acc = wmma_bf16(a, b, acc);
    }
    int nc = wv * 16 + (lane & 15), hi = lane >> 4;
#pragma unroll
    for (int r = 0; r < 8; ++r) xs[(r + hi * 8) * 132 + nc] += acc[r];
  }
  __syncthreads();
  // ff layernorm -> sa
  {
    int row = wv * 2 + (lane >> 4);
    int cl = lane & 15;
    float s = 0.f, s2 = 0.f, vals[8];
#pragma unroll
    for (int t = 0; t < 8; ++t) { float v = xs[row * 132 + cl + 16 * t]; vals[t] = v; s += v; s2 += v * v; }
#pragma unroll
    for (int m = 1; m < 16; m <<= 1) { s += __shfl_xor(s, m, 32); s2 += __shfl_xor(s2, m, 32); }
    float mu = s * (1.f / 128.f);
    float var = s2 * (1.f / 128.f) - mu * mu;
    float inv = rsqrtf(var + 1e-6f);
#pragma unroll
    for (int t = 0; t < 8; ++t) {
      int c = cl + 16 * t;
      sa[row * 136 + c] = (bf16_t)((vals[t] - mu) * inv * flnw[c] + flnb[c]);
    }
  }
  __syncthreads();
  // h2 = relu(sa @ wi^T) : 32 tiles
  for (int tt = 0; tt < 4; ++tt) {
    int tile = wv + tt * 8;
    v8f acc = zero8();
#pragma unroll
    for (int kk = 0; kk < 4; ++kk) {
      v16bf a = load_a(sa, 136, kk * 32, lane);
      v16bf b = load_b(wi, 128, tile * 16, kk * 32, lane);
      acc = wmma_bf16(a, b, acc);
    }
    int nc = tile * 16 + (lane & 15), hi = lane >> 4;
#pragma unroll
    for (int r = 0; r < 8; ++r) {
      float v = acc[r];
      h2[(r + hi * 8) * 528 + nc] = (bf16_t)(v > 0.f ? v : 0.f);
    }
  }
  __syncthreads();
  // x = xs + h2 @ wo^T
  {
    v8f acc = zero8();
#pragma unroll
    for (int kk = 0; kk < 16; ++kk) {
      v16bf a = load_a(h2, 528, kk * 32, lane);
      v16bf b = load_b(wo, 512, wv * 16, kk * 32, lane);
      acc = wmma_bf16(a, b, acc);
    }
    int nc = wv * 16 + (lane & 15), hi = lane >> 4;
#pragma unroll
    for (int r = 0; r < 8; ++r) {
      int m = r + hi * 8;
      x[(n0 + m) * D_ + nc] = xs[m * 132 + nc] + acc[r];
    }
  }
}

// ---------------- host ----------------
static inline size_t align256(size_t v) { return (v + 255) & ~(size_t)255; }

extern "C" void kernel_launch(void* const* d_in, const int* in_sizes, int n_in,
                              void* d_out, int out_size, void* d_ws, size_t ws_size,
                              hipStream_t stream) {
  (void)in_sizes; (void)n_in; (void)out_size; (void)ws_size;
  const int*   node_ids = (const int*)d_in[0];
  const int*   edge_index = (const int*)d_in[1];
  const int*   edge_ids = (const int*)d_in[2];
  const float* node_emb = (const float*)d_in[3];
  const float* edge_emb = (const float*)d_in[4];
  const float* ln_w  = (const float*)d_in[5];
  const float* ln_b  = (const float*)d_in[6];
  const float* q_w   = (const float*)d_in[7];
  const float* relA_w = (const float*)d_in[8];
  const float* relA_b = (const float*)d_in[9];
  const float* relB_w = (const float*)d_in[10];
  const float* relB_b = (const float*)d_in[11];
  const float* relG_w = (const float*)d_in[12];
  const float* relG_b = (const float*)d_in[13];
  const float* k_w   = (const float*)d_in[14];
  const float* v_w   = (const float*)d_in[15];
  const float* o_w   = (const float*)d_in[16];
  const float* ff_ln_w = (const float*)d_in[17];
  const float* ff_ln_b = (const float*)d_in[18];
  const float* wi_w  = (const float*)d_in[19];
  const float* wo_w  = (const float*)d_in[20];
  const int* src = edge_index;
  const int* dst = edge_index + E_;

  char* ws = (char*)d_ws;
  size_t off = 0;
  auto alloc = [&](size_t bytes) -> char* {
    char* p = ws + off; off = align256(off + bytes); return p;
  };
  float*  X    = (float*)alloc((size_t)N_ * D_ * 4);
  float*  Q    = (float*)alloc((size_t)N_ * D_ * 4);
  bf16_t* EF   = (bf16_t*)alloc((size_t)E_ * D_ * 2);
  bf16_t* VB   = (bf16_t*)alloc((size_t)E_ * D_ * 2);
  float*  SC   = (float*)alloc((size_t)E_ * H_ * 4);
  float*  SMAX = (float*)alloc((size_t)N_ * H_ * 4);
  float*  DEN  = (float*)alloc((size_t)N_ * H_ * 4);
  float*  AGG  = (float*)alloc((size_t)N_ * D_ * 4);
  bf16_t* QW = (bf16_t*)alloc((size_t)L_ * 128 * 128 * 2);
  bf16_t* RA = (bf16_t*)alloc((size_t)L_ * 512 * 256 * 2);
  bf16_t* RB = (bf16_t*)alloc((size_t)L_ * 128 * 512 * 2);
  bf16_t* RG = (bf16_t*)alloc((size_t)L_ * 128 * 512 * 2);
  bf16_t* KW = (bf16_t*)alloc((size_t)L_ * 128 * 256 * 2);
  bf16_t* VW = (bf16_t*)alloc((size_t)L_ * 128 * 128 * 2);
  bf16_t* OW = (bf16_t*)alloc((size_t)L_ * 128 * 128 * 2);
  bf16_t* WI = (bf16_t*)alloc((size_t)L_ * 512 * 128 * 2);
  bf16_t* WO = (bf16_t*)alloc((size_t)L_ * 128 * 512 * 2);

  auto cvt = [&](const float* in, bf16_t* out, int n) {
    k_cvt<<<(n + 255) / 256, 256, 0, stream>>>(in, out, n);
  };
  cvt(q_w,   QW, L_ * 128 * 128);
  cvt(relA_w, RA, L_ * 512 * 256);
  cvt(relB_w, RB, L_ * 128 * 512);
  cvt(relG_w, RG, L_ * 128 * 512);
  cvt(k_w,   KW, L_ * 128 * 256);
  cvt(v_w,   VW, L_ * 128 * 128);
  cvt(o_w,   OW, L_ * 128 * 128);
  cvt(wi_w,  WI, L_ * 512 * 128);
  cvt(wo_w,  WO, L_ * 128 * 512);

  k_gather_x<<<(N_ * D_ + 255) / 256, 256, 0, stream>>>(node_ids, node_emb, X, N_ * D_);
  k_gather_e<<<(E_ * D_ + 255) / 256, 256, 0, stream>>>(edge_ids, edge_emb, EF, E_ * D_);

  for (int l = 0; l < L_; ++l) {
    k_q<<<N_ / 16, 256, 0, stream>>>(X, ln_w + l * D_, ln_b + l * D_,
                                     QW + l * 128 * 128, Q);
    k_init<<<(N_ * D_ + 255) / 256, 256, 0, stream>>>(SMAX, DEN, AGG, N_ * H_, N_ * D_);
    k_edge<<<E_ / EB, 256, SMEM_EDGE, stream>>>(
        X, EF, Q, src, dst,
        RA + l * 512 * 256, relA_b + l * FF_,
        RG + l * 128 * 512, relG_b + l * D_,
        RB + l * 128 * 512, relB_b + l * D_,
        KW + l * 128 * 256, VW + l * 128 * 128,
        VB, SC, SMAX);
    k_soft<<<(E_ * H_ + 255) / 256, 256, 0, stream>>>(dst, SC, SMAX, DEN, E_ * H_);
    k_scatter<<<(E_ * D_ + 255) / 256, 256, 0, stream>>>(dst, SC, DEN, VB, AGG, E_ * D_);
    k_node<<<N_ / 16, 256, 0, stream>>>(X, AGG, OW + l * 128 * 128,
                                        ff_ln_w + l * D_, ff_ln_b + l * D_,
                                        WI + l * 512 * 128, WO + l * 128 * 512);
  }
  k_copy<<<(N_ * D_ + 255) / 256, 256, 0, stream>>>(X, (float*)d_out, N_ * D_);
}
